// MoreMultiDynamicWeights_87041807221281
// MI455X (gfx1250) — matrix operand order repaired and checked by reading, not compiled, not run on hardware
//
#include <hip/hip_runtime.h>
#include <hip/hip_bf16.h>

typedef __attribute__((ext_vector_type(16))) _Float16 v16h;
typedef __attribute__((ext_vector_type(8)))  float    v8f;

#define P_TOT 18432      // total pixels across batch (2*96*96)
#define CCH   256        // channels of x
#define K1    1536       // concat channels (6*256)
#define HW    9216       // 96*96
#define WID   96

// k_guid LDS tile: 48 rows x 512B data, padded
#define GROWB 528
#define GROWH 264
#define GBUFB (48 * GROWB)     // 25,344 B per buffer
// k_gemm LDS tile: 64 rows x 256B data (K-chunk = 128), padded
#define WROWB 272
#define WROWH 136
#define WBUFB (64 * WROWB)     // 17,408 B per buffer

union V16 { int4 q[2]; v16h h; };
union H2U { unsigned u; _Float16 h[2]; };

__device__ __forceinline__ int4 ld_int4(const void* p) {
  return *reinterpret_cast<const int4*>(p);
}

// async copy of one 16B segment global -> LDS (GVS addressing), ASYNCcnt-tracked
__device__ __forceinline__ void async_cp16(unsigned ldsOff, unsigned gOff,
                                           const void* base) {
  asm volatile("global_load_async_to_lds_b128 %0, %1, %2"
               :: "v"(ldsOff), "v"(gOff), "s"(base)
               : "memory");
}
__device__ __forceinline__ void async_wait0() {
  asm volatile("s_wait_asynccnt 0" ::: "memory");
}

// ---------- prep: x (NCHW f32) -> cat[:, 0:256] (pixel-major f16) ----------
__global__ __launch_bounds__(256) void k_prep_x(const float* __restrict__ x,
                                                _Float16* __restrict__ cat) {
  const size_t idx = (size_t)blockIdx.x * 256 + threadIdx.x; // P_TOT*CCH
  const int c = (int)(idx & 255);
  const int p = (int)(idx >> 8);
  const int n = p / HW;
  const int hw = p - n * HW;
  const float v = x[((size_t)(n * CCH + c)) * HW + hw];
  cat[(size_t)p * K1 + c] = (_Float16)v;
}

// ---------- prep: W_cat (45,256,3,3) -> bw[tap][oc(48)][c] f16 ----------
__global__ __launch_bounds__(256) void k_prep_bw(const float* __restrict__ wcat,
                                                 _Float16* __restrict__ bw) {
  const int idx = blockIdx.x * 256 + threadIdx.x;  // 9*48*256
  const int c = idx & 255;
  const int oc = (idx >> 8) % 48;
  const int tap = idx / (48 * 256);
  float v = 0.f;
  if (oc < 45) v = wcat[((size_t)oc * CCH + c) * 9 + tap];
  bw[idx] = (_Float16)v;
}

// ---------- prep: W_scale (256,1536) f32 -> f16 ----------
__global__ __launch_bounds__(256) void k_prep_wh(const float* __restrict__ ws,
                                                 _Float16* __restrict__ wh) {
  const int idx = blockIdx.x * 256 + threadIdx.x;  // 256*1536
  wh[idx] = (_Float16)ws[idx];
}

// ---------- guidance conv 3x3 as WMMA GEMM: M=16px, N=48oc, K=9*256 ----------
// Weights double-buffered in LDS: stage tap t+1 (async) while computing tap t.
__global__ __launch_bounds__(256) void k_guid(const _Float16* __restrict__ cat,
                                              const _Float16* __restrict__ bw,
                                              float* __restrict__ guid) {
  __shared__ _Float16 sB[2][48 * GROWH];   // 2 x 25,344 B

  const int tid   = threadIdx.x;
  const int lane  = tid & 31;
  const int wv    = tid >> 5;
  const int ptile = blockIdx.x * 8 + wv;    // 1152 p-tiles, 144 blocks * 8 waves
  const int laneN = lane & 15;
  const int hi    = lane >> 4;
  const int pix   = ptile * 16 + laneN;     // this lane's A row (pixel)
  const int n  = pix / HW;
  const int hw = pix - n * HW;
  const int h  = hw / WID;
  const int w  = hw - h * WID;

  v8f acc[3] = {};
  const unsigned char* catB = (const unsigned char*)cat;
  const unsigned char* sBB  = (const unsigned char*)&sB[0][0];
  const unsigned sBase = (unsigned)(uintptr_t)&sB[0][0]; // flat->LDS low 32 bits

  // prologue: stage tap 0 into buffer 0
  for (int s = tid; s < 1536; s += 256) {      // 1536 segments of 16B
    const int co = s >> 5;                     // 32 segs per row
    const int kk = s & 31;
    async_cp16(sBase + co * GROWB + kk * 16,
               (unsigned)(co * 512 + kk * 16), bw);
  }
  async_wait0();
  __syncthreads();

  for (int tap = 0; tap < 9; ++tap) {
    const unsigned bufOff = (unsigned)(tap & 1) * GBUFB;
    // ---- stage next tap into the other buffer (overlaps compute) ----
    if (tap < 8) {
      const unsigned nxtOff = (unsigned)((tap + 1) & 1) * GBUFB;
      for (int s = tid; s < 1536; s += 256) {
        const int co = s >> 5;
        const int kk = s & 31;
        async_cp16(sBase + nxtOff + co * GROWB + kk * 16,
                   (unsigned)(((tap + 1) * 48 + co) * 512 + kk * 16), bw);
      }
    }

    const int dy = tap / 3 - 1;
    const int dx = tap % 3 - 1;
    const bool valid = ((unsigned)(h + dy) < WID) && ((unsigned)(w + dx) < WID);
    const long q = (long)pix + (long)dy * WID + dx;
    for (int ks = 0; ks < 8; ++ks) {    // K = 256 per tap, 32 per step
      V16 a;
      a.q[0] = make_int4(0, 0, 0, 0);
      a.q[1] = make_int4(0, 0, 0, 0);
      if (valid) {
        const unsigned char* ap = catB + q * (long)(K1 * 2) + ks * 64 + hi * 16;
        a.q[0] = ld_int4(ap);
        a.q[1] = ld_int4(ap + 32);
      }
#pragma unroll
      for (int nt = 0; nt < 3; ++nt) {
        const unsigned char* bp =
            sBB + bufOff + (nt * 16 + laneN) * GROWB + ks * 64 + hi * 16;
        V16 b;
        b.q[0] = ld_int4(bp);
        b.q[1] = ld_int4(bp + 32);
        acc[nt] = __builtin_amdgcn_wmma_f32_16x16x32_f16(
            false, a.h, false, b.h, (short)0, acc[nt], false, false);
      }
    }
    async_wait0();
    __syncthreads();   // staged buffer ready; all reads of bufOff complete
  }
#pragma unroll
  for (int nt = 0; nt < 3; ++nt) {
    const int oc = nt * 16 + laneN;
    if (oc < 45) {
      float* gp = guid + (size_t)oc * P_TOT + ptile * 16 + hi * 8;
      *(float4*)gp       = make_float4(acc[nt][0], acc[nt][1], acc[nt][2], acc[nt][3]);
      *(float4*)(gp + 4) = make_float4(acc[nt][4], acc[nt][5], acc[nt][6], acc[nt][7]);
    }
  }
}

// ---------- softmax + 5 dynamic dilated branches -> cat[:, 256:1536] ----------
__global__ __launch_bounds__(128) void k_dyn(_Float16* cat,
                                             const float* __restrict__ guid) {
  const int p  = blockIdx.x;
  const int br = blockIdx.y;     // 0..4
  const int t  = threadIdx.x;    // handles channels 2t, 2t+1
  const int dils[5] = {1, 6, 12, 24, 36};
  const int d = dils[br];

  __shared__ float g9[9];
  if (t < 9) g9[t] = guid[(size_t)(9 * br + t) * P_TOT + p];
  __syncthreads();

  float m = g9[0];
#pragma unroll
  for (int k = 1; k < 9; ++k) m = fmaxf(m, g9[k]);
  float e[9], s = 0.f;
#pragma unroll
  for (int k = 0; k < 9; ++k) { e[k] = __expf(g9[k] - m); s += e[k]; }
  const float inv = 1.f / s;

  const int n  = p / HW;
  const int hw = p - n * HW;
  const int h  = hw / WID;
  const int w  = hw - h * WID;

  const unsigned* catU = (const unsigned*)cat;  // row stride K1/2 = 768 dwords
  float a0 = 0.f, a1 = 0.f;
  int k = 0;
#pragma unroll
  for (int iy = -1; iy <= 1; ++iy) {
#pragma unroll
    for (int ix = -1; ix <= 1; ++ix, ++k) {
      const int dy = iy * d, dx = ix * d;
      const float wgt = e[k] * inv;
      if (((unsigned)(h + dy) < WID) && ((unsigned)(w + dx) < WID)) {
        const long q = (long)p + (long)dy * WID + dx;
        H2U cv; cv.u = catU[q * (K1 / 2) + t];
        a0 += wgt * (float)cv.h[0];
        a1 += wgt * (float)cv.h[1];
      }
    }
  }
  H2U o; o.h[0] = (_Float16)a0; o.h[1] = (_Float16)a1;
  ((unsigned*)cat)[(size_t)p * (K1 / 2) + (br + 1) * 128 + t] = o.u;
}

// ---------- 1x1 conv as WMMA GEMM: y[co][p] = sum_k W[co][k]*cat[p][k] ----------
// Weights double-buffered in LDS (K-chunks of 128); A pipelined from global.
__global__ __launch_bounds__(256) void k_gemm(const _Float16* __restrict__ cat,
                                              const _Float16* __restrict__ wh,
                                              float* __restrict__ y) {
  __shared__ _Float16 sW[2][64 * WROWH];   // 2 x 17,408 B

  const int tid    = threadIdx.x;
  const int lane   = tid & 31;
  const int wv     = tid >> 5;
  const int ptile  = blockIdx.x * 8 + wv;   // 1152 p-tiles
  const int laneN  = lane & 15;
  const int hi     = lane >> 4;
  const int coBase = blockIdx.y * 64;       // 4 groups of 64 out channels
  const int pix    = ptile * 16 + laneN;

  v8f acc[4] = {};
  const unsigned char* arow = (const unsigned char*)cat + (size_t)pix * (K1 * 2);
  const unsigned char* whBlk =
      (const unsigned char*)wh + (size_t)coBase * (K1 * 2);
  const unsigned char* sWB = (const unsigned char*)&sW[0][0];
  const unsigned sBase = (unsigned)(uintptr_t)&sW[0][0];

  // prologue: stage chunk 0 into buffer 0 (64 rows x 256B)
  for (int s = tid; s < 1024; s += 256) {    // 1024 segments of 16B
    const int co = s >> 4;                   // 16 segs per row
    const int kk = s & 15;
    async_cp16(sBase + co * WROWB + kk * 16,
               (unsigned)(co * (K1 * 2) + kk * 16), whBlk);
  }
  async_wait0();
  __syncthreads();

  for (int ck = 0; ck < 12; ++ck) {    // K chunks of 128 (256B per row)
    const unsigned bufOff = (unsigned)(ck & 1) * WBUFB;
    // ---- stage next chunk into the other buffer (overlaps compute) ----
    if (ck < 11) {
      const unsigned nxtOff = (unsigned)((ck + 1) & 1) * WBUFB;
      for (int s = tid; s < 1024; s += 256) {
        const int co = s >> 4;
        const int kk = s & 15;
        async_cp16(sBase + nxtOff + co * WROWB + kk * 16,
                   (unsigned)(co * (K1 * 2) + (ck + 1) * 256 + kk * 16), whBlk);
      }
    }

    // ---- pipelined A loads + WMMA over 4 k-steps ----
    const unsigned char* ap0 = arow + ck * 256 + hi * 16;
    V16 a;
    a.q[0] = ld_int4(ap0);
    a.q[1] = ld_int4(ap0 + 32);
    for (int ks = 0; ks < 4; ++ks) {
      V16 an = a;
      if (ks < 3) {
        const unsigned char* ap = ap0 + (ks + 1) * 64;
        an.q[0] = ld_int4(ap);
        an.q[1] = ld_int4(ap + 32);
      }
#pragma unroll
      for (int nt = 0; nt < 4; ++nt) {
        const unsigned char* bp =
            sWB + bufOff + (nt * 16 + laneN) * WROWB + ks * 64 + hi * 16;
        V16 b;
        b.q[0] = ld_int4(bp);
        b.q[1] = ld_int4(bp + 32);
        acc[nt] = __builtin_amdgcn_wmma_f32_16x16x32_f16(
            false, a.h, false, b.h, (short)0, acc[nt], false, false);
      }
      a = an;
    }
    async_wait0();
    __syncthreads();   // staged buffer ready; all reads of bufOff complete
  }

#pragma unroll
  for (int nt = 0; nt < 4; ++nt) {
    const int co = coBase + nt * 16 + laneN;
    float* yp = y + (size_t)co * P_TOT + ptile * 16 + hi * 8;
    *(float4*)yp       = make_float4(acc[nt][0], acc[nt][1], acc[nt][2], acc[nt][3]);
    *(float4*)(yp + 4) = make_float4(acc[nt][4], acc[nt][5], acc[nt][6], acc[nt][7]);
  }
}

// ---------- BN stats: per-channel mean + inv-std over 18432 elems ----------
__global__ __launch_bounds__(256) void k_stats(const float* __restrict__ y,
                                               float* __restrict__ stats) {
  const int c = blockIdx.x;
  const int t = threadIdx.x;
  const float* row = y + (size_t)c * P_TOT;
  float s = 0.f, ss = 0.f;
  for (int i = t; i < P_TOT; i += 256) { float v = row[i]; s += v; ss += v * v; }
  __shared__ float sh[256], sh2[256];
  sh[t] = s; sh2[t] = ss;
  __syncthreads();
  for (int o = 128; o > 0; o >>= 1) {
    if (t < o) { sh[t] += sh[t + o]; sh2[t] += sh2[t + o]; }
    __syncthreads();
  }
  if (t == 0) {
    const float mean = sh[0] * (1.f / P_TOT);
    const float var  = sh2[0] * (1.f / P_TOT) - mean * mean;
    stats[c]       = mean;
    stats[256 + c] = rsqrtf(var + 1e-5f);
  }
}

// ---------- BN apply: y[c][p] -> out NCHW ----------
__global__ __launch_bounds__(256) void k_bn(const float* __restrict__ y,
                                            const float* __restrict__ stats,
                                            const float* __restrict__ gamma,
                                            const float* __restrict__ beta,
                                            float* __restrict__ out) {
  const size_t idx = (size_t)blockIdx.x * 256 + threadIdx.x;  // 2*256*9216
  const int hw = (int)(idx % HW);
  const int c  = (int)((idx / HW) & 255);
  const int n  = (int)(idx / ((size_t)HW * 256));
  const int p  = n * HW + hw;
  const float v = y[(size_t)c * P_TOT + p];
  out[idx] = (v - stats[c]) * stats[256 + c] * gamma[c] + beta[c];
}

extern "C" void kernel_launch(void* const* d_in, const int* in_sizes, int n_in,
                              void* d_out, int out_size, void* d_ws, size_t ws_size,
                              hipStream_t stream) {
  (void)in_sizes; (void)n_in; (void)out_size; (void)ws_size;
  const float* x      = (const float*)d_in[0];
  const float* Wcat   = (const float*)d_in[1];
  const float* Wscale = (const float*)d_in[2];
  const float* gamma  = (const float*)d_in[3];
  const float* beta   = (const float*)d_in[4];

  unsigned char* ws = (unsigned char*)d_ws;
  _Float16* cat  = (_Float16*)(ws + 0);          // 18432*1536*2 = 56,623,104
  float*    guid = (float*)(ws + 56623104);      // 48*18432*4   =  3,538,944
  _Float16* bw   = (_Float16*)(ws + 60162048);   // 9*48*256*2   =    221,184
  _Float16* wh   = (_Float16*)(ws + 60383232);   // 256*1536*2   =    786,432
  float*    y    = (float*)(ws + 61169664);      // 256*18432*4  = 18,874,368
  float*    st   = (float*)(ws + 80044032);      // 512*4

  k_prep_x<<<18432, 256, 0, stream>>>(x, cat);
  k_prep_bw<<<432, 256, 0, stream>>>(Wcat, bw);
  k_prep_wh<<<1536, 256, 0, stream>>>(Wscale, wh);
  k_guid<<<144, 256, 0, stream>>>(cat, bw, guid);
  dim3 gdyn(P_TOT, 5);
  k_dyn<<<gdyn, 128, 0, stream>>>(cat, guid);
  dim3 gg(144, 4);
  k_gemm<<<gg, 256, 0, stream>>>(cat, wh, y);
  k_stats<<<256, 256, 0, stream>>>(y, st);
  k_bn<<<18432, 256, 0, stream>>>(y, st, gamma, beta, (float*)d_out);
}
